// LinearAutoregressiveHMM_47545287967191
// MI455X (gfx1250) — matrix operand order repaired and checked by reading, not compiled, not run on hardware
//
#include <hip/hip_runtime.h>
#include <cmath>

// CDNA5 / gfx1250 — wave32, WMMA f32<-f16 16x16x32.
typedef __attribute__((ext_vector_type(16))) _Float16 v16h;
typedef __attribute__((ext_vector_type(8)))  float    v8f;

#define NB 8192   // B
#define LL 64     // L
#define CC 4      // C
#define KK 8      // K
#define PP 16     // P
#define FF 256    // L*C (GEMM reduction dim)
#define JJ 32     // K*C (GEMM output cols)
#define WPITCH 264  // padded f16 pitch for W rows (264*2B = 528B -> conflict-free column reads)

__global__ __launch_bounds__(128) void hmm_fused_kernel(
    const float* __restrict__ em,     // (B,L,C)
    const float* __restrict__ trans,  // (K,K)
    const float* __restrict__ initd,  // (K)
    const float* __restrict__ means,  // (K,C)
    const float* __restrict__ covch,  // (K,C)
    const float* __restrict__ W,      // (K,C,F) flat: row j=k*C+c, 256 wide
    float* __restrict__ out)          // (B,P,C)
{
    __shared__ _Float16 sW[JJ * WPITCH];          // W^T rows (j, f) in f16, padded pitch
    __shared__ _Float16 sA[2 * FF];               // padded emission window, f16 (A source)
    __shared__ float    sX[2 * FF + PP * CC];     // [256 zeros | 256 emissions | 64 preds]
    __shared__ float    sAR[LL * JJ];             // ar (64 x 32)
    __shared__ float    sELP[LL * KK];            // emission log-probs (64 x 8)
    __shared__ int      sState;

    const int tid  = threadIdx.x;
    const int lane = tid & 31;
    const int wv   = tid >> 5;          // wave 0..3
    const int b    = blockIdx.x;

    // hint the L2: W is reused by every block
    __builtin_prefetch(W + tid * 64, 0, 1);

    // ---- stage W as f16 with padded pitch (8192 elems, 64/thread) ----
    for (int idx = tid; idx < KK * CC * FF; idx += 128) {
        const int j = idx >> 8, f = idx & 255;
        sW[j * WPITCH + f] = (_Float16)W[idx];
    }
    // ---- stage padded emission window (zeros | emissions), f32 + f16 ----
    for (int i = tid; i < FF; i += 128) {
        const float v = em[(size_t)b * FF + i];
        sX[i] = 0.0f;            sX[FF + i] = v;
        sA[i] = (_Float16)0.0f;  sA[FF + i] = (_Float16)v;
    }
    __syncthreads();

    // ================= WMMA GEMM: A(64x256, Hankel) x W^T(256x32) =================
    // wave wv owns rows t = 16*wv .. 16*wv+15; two 16-col N-tiles.
    {
        const int g = lane >> 4;           // half-wave group (ISA 16-bit frag layout)
        const int t = wv * 16 + (lane & 15);
        const int n = lane & 15;
        v8f c0 = {}; v8f c1 = {};
        for (int kb = 0; kb < FF; kb += 32) {
            v16h a, b0, b1;
            // A[t, f] = sA[t*4 + f]; halves 0..7 <- K kb+8g+q, 8..15 <- K kb+16+8g+q
            const int abase = t * 4 + kb + 8 * g;
            #pragma unroll
            for (int q = 0; q < 8; ++q) {
                a[q]     = sA[abase + q];
                a[q + 8] = sA[abase + 16 + q];
            }
            // B[f, j] = sW[j*WPITCH + f]; halves 0..7 <- K kb+8g+q, 8..15 <- K kb+16+8g+q, column n (and n+16)
            const int bbase = kb + 8 * g;
            #pragma unroll
            for (int q = 0; q < 16; ++q) {
                b0[q] = sW[n * WPITCH + bbase + q + (q >= 8 ? 8 : 0)];
                b1[q] = sW[(n + 16) * WPITCH + bbase + q + (q >= 8 ? 8 : 0)];
            }
            c0 = __builtin_amdgcn_wmma_f32_16x16x32_f16(false, a, false, b0,
                                                        (short)0, c0, false, false);
            c1 = __builtin_amdgcn_wmma_f32_16x16x32_f16(false, a, false, b1,
                                                        (short)0, c1, false, false);
        }
        // C layout: VGPR r -> M = r + 8*g, N = lane&15
        #pragma unroll
        for (int r = 0; r < 8; ++r) {
            const int tt = wv * 16 + r + 8 * g;
            sAR[tt * JJ + n]      = c0[r];
            sAR[tt * JJ + n + 16] = c1[r];
        }
    }
    __syncthreads();

    // ================= emission log-probabilities =================
    const float LOG2PI = 1.8378770664093453f;
    for (int idx = tid; idx < LL * KK; idx += 128) {
        const int t = idx >> 3, k = idx & 7;
        float maha = 0.f, logdet = 0.f;
        #pragma unroll
        for (int c = 0; c < CC; ++c) {
            const float var = expf(covch[k * CC + c]) + 1e-6f;
            const float d = sX[FF + t * CC + c]
                          - (means[k * CC + c] + sAR[t * JJ + k * CC + c]);
            maha   += d * d / var;
            logdet += logf(var);
        }
        sELP[t * KK + k] = -0.5f * (maha + logdet + CC * LOG2PI);
    }
    __syncthreads();

    // ================= HMM forward scan (wave 0, width-8 shuffles) =================
    if (wv == 0) {
        const int k = lane & 7;
        float lt[KK];
        #pragma unroll
        for (int i = 0; i < KK; ++i) {  // log(softmax(trans,axis=1) + 1e-8), column k
            float m = -1e30f, s = 0.f;
            #pragma unroll
            for (int j = 0; j < KK; ++j) m = fmaxf(m, trans[i * KK + j]);
            #pragma unroll
            for (int j = 0; j < KK; ++j) s += expf(trans[i * KK + j] - m);
            lt[i] = logf(expf(trans[i * KK + k] - m) / s + 1e-8f);
        }
        float m0 = -1e30f, s0 = 0.f;
        #pragma unroll
        for (int j = 0; j < KK; ++j) m0 = fmaxf(m0, initd[j]);
        #pragma unroll
        for (int j = 0; j < KK; ++j) s0 += expf(initd[j] - m0);
        float la = (initd[k] - m0 - logf(s0)) + sELP[k];

        for (int t = 1; t < LL; ++t) {
            float v[KK];
            #pragma unroll
            for (int i = 0; i < KK; ++i) v[i] = __shfl(la, i, KK) + lt[i];
            float m = -1e30f;
            #pragma unroll
            for (int i = 0; i < KK; ++i) m = fmaxf(m, v[i]);
            float s = 0.f;
            #pragma unroll
            for (int i = 0; i < KK; ++i) s += expf(v[i] - m);
            la = m + logf(s) + sELP[t * KK + k];
        }
        // argmax over states (first-max, matching jnp.argmax)
        float best = -1e30f; int bk = 0;
        #pragma unroll
        for (int i = 0; i < KK; ++i) {
            const float v = __shfl(la, i, KK);
            if (v > best) { best = v; bk = i; }
        }
        if (tid == 0) sState = bk;
    }
    __syncthreads();

    // ================= autoregressive prediction (P steps) =================
    // state chain is prediction-independent; each thread tracks it locally.
    int st = sState;
    const int c = wv;  // one output channel per wave
    for (int p = 0; p < PP; ++p) {
        // advance state: argmax of raw row == argmax of softmax row
        float best = -1e30f; int ns = 0;
        #pragma unroll
        for (int j = 0; j < KK; ++j) {
            const float v = trans[st * KK + j];
            if (v > best) { best = v; ns = j; }
        }
        st = ns;
        const float* Wr = W + (size_t)st * CC * FF + (size_t)c * FF;
        float part = 0.f;
        #pragma unroll
        for (int f = lane; f < FF; f += 32)
            part += sX[FF + p * CC + f] * Wr[f];   // sliding window = moving LDS base
        #pragma unroll
        for (int m = 16; m; m >>= 1) part += __shfl_xor(part, m);
        if (lane == 0) {
            const float pr = part + means[st * CC + c];
            sX[2 * FF + p * CC + c] = pr;                         // feed back
            out[(size_t)b * (PP * CC) + p * CC + c] = pr;         // (B,P,C)
        }
        __syncthreads();
    }
}

extern "C" void kernel_launch(void* const* d_in, const int* in_sizes, int n_in,
                              void* d_out, int out_size, void* d_ws, size_t ws_size,
                              hipStream_t stream) {
    (void)in_sizes; (void)n_in; (void)out_size; (void)d_ws; (void)ws_size;
    hmm_fused_kernel<<<NB, 128, 0, stream>>>(
        (const float*)d_in[0],   // emissions
        (const float*)d_in[1],   // transition_matrix
        (const float*)d_in[2],   // initial_distribution
        (const float*)d_in[3],   // means
        (const float*)d_in[4],   // covar_chol
        (const float*)d_in[5],   // W_stack
        (float*)d_out);
}
